// BaseWeeklyBank_13950053777871
// MI455X (gfx1250) — matrix operand order repaired and checked by reading, not compiled, not run on hardware
//
#include <hip/hip_runtime.h>

typedef __attribute__((ext_vector_type(16))) _Float16 v16h;
typedef __attribute__((ext_vector_type(8)))  float    v8f;
typedef __attribute__((ext_vector_type(4)))  float    v4f;
typedef __attribute__((ext_vector_type(4)))  unsigned int v4u;
typedef __attribute__((ext_vector_type(8)))  int      v8i;
typedef __attribute__((ext_vector_type(4)))  int      v4i;

#define N_NODES 400
#define N_EDGES 3200
#define N_TOT   3600   // edges + self loops
#define T_STEPS 2016

// LDS byte offsets for the temporal kernel (see layout in k_temporal)
#define LDS_W_OFF 25600          // byte offset of sW1 (TDM destination)
#define W_COMB_ELEMS (32 * 256 + 64 * 64)   // 12288 f16 elements

// ---------------------------------------------------------------------------
// CSR build (deterministic, single thread; edge order: original edges then
// self-loop, matching the reference's concatenate([edge_index, loop])).
// ---------------------------------------------------------------------------
__global__ void k_csr(const int* __restrict__ ei,
                      int* __restrict__ off, int* __restrict__ lst) {
    if (threadIdx.x != 0 || blockIdx.x != 0) return;
    int cnt[N_NODES];
    for (int n = 0; n < N_NODES; ++n) cnt[n] = 0;
    for (int e = 0; e < N_TOT; ++e) {
        int d = (e < N_EDGES) ? ei[N_EDGES + e] : (e - N_EDGES);
        cnt[d]++;
    }
    off[0] = 0;
    for (int n = 0; n < N_NODES; ++n) off[n + 1] = off[n] + cnt[n];
    for (int n = 0; n < N_NODES; ++n) cnt[n] = off[n];
    for (int e = 0; e < N_TOT; ++e) {
        int s, d;
        if (e < N_EDGES) { s = ei[e]; d = ei[N_EDGES + e]; }
        else             { s = e - N_EDGES; d = s; }
        lst[cnt[d]++] = s;
    }
}

// ---------------------------------------------------------------------------
// Small dense helpers for the static encoder / calendar path
// ---------------------------------------------------------------------------
__global__ void k_dense(const float* __restrict__ x, const float* __restrict__ W,
                        float* __restrict__ out, int n, int fin, int fout) {
    int tid = blockIdx.x * blockDim.x + threadIdx.x;
    if (tid >= n * fout) return;
    int i = tid / fout, j = tid % fout;
    float acc = 0.f;
    for (int k = 0; k < fin; ++k) acc += x[i * fin + k] * W[k * fout + j];
    out[tid] = acc;
}

__global__ void k_al(const float* __restrict__ h,
                     const float* __restrict__ a_src, const float* __restrict__ a_dst,
                     float* __restrict__ als, float* __restrict__ ald,
                     int n, int H, int C) {
    int tid = blockIdx.x * blockDim.x + threadIdx.x;
    if (tid >= n * H) return;
    int i = tid / H, hh = tid % H;
    const float* hr = h + (size_t)i * H * C + hh * C;
    float s = 0.f, d = 0.f;
    for (int c = 0; c < C; ++c) {
        float v = hr[c];
        s += v * a_src[hh * C + c];
        d += v * a_dst[hh * C + c];
    }
    als[tid] = s; ald[tid] = d;
}

// Deterministic per-node GAT aggregation (mean over heads + bias + relu)
__global__ void k_gat_agg(const float* __restrict__ h,
                          const float* __restrict__ als, const float* __restrict__ ald,
                          const int* __restrict__ off, const int* __restrict__ lst,
                          const float* __restrict__ bias, float* __restrict__ out,
                          int n, int H, int C) {
    int i = blockIdx.x * blockDim.x + threadIdx.x;
    if (i >= n) return;
    float acc[32];
    for (int c = 0; c < C; ++c) acc[c] = 0.f;
    int b = off[i], e = off[i + 1];
    for (int hh = 0; hh < H; ++hh) {
        float ad = ald[i * H + hh];
        float m = -1e30f;
        for (int k = b; k < e; ++k) {
            float v = als[lst[k] * H + hh] + ad;
            v = v > 0.f ? v : 0.2f * v;
            m = fmaxf(m, v);
        }
        float den = 0.f;
        for (int k = b; k < e; ++k) {
            float v = als[lst[k] * H + hh] + ad;
            v = v > 0.f ? v : 0.2f * v;
            den += __expf(v - m);
        }
        float inv = 1.f / den;
        for (int k = b; k < e; ++k) {
            float v = als[lst[k] * H + hh] + ad;
            v = v > 0.f ? v : 0.2f * v;
            float w = __expf(v - m) * inv;
            const float* hs = h + (size_t)lst[k] * H * C + hh * C;
            for (int c = 0; c < C; ++c) acc[c] += w * hs[c];
        }
    }
    float sc = 1.f / (float)H;
    for (int c = 0; c < C; ++c) {
        float y = acc[c] * sc + bias[c];
        out[i * C + c] = y > 0.f ? y : 0.f;
    }
}

// Calendar encoder: [7,288] grid -> cal[2016,32]
__global__ void k_cal(const float* __restrict__ wd_emb, const float* __restrict__ sl_emb,
                      const float* __restrict__ Wc1, const float* __restrict__ bc1,
                      const float* __restrict__ Wc2, const float* __restrict__ bc2,
                      float* __restrict__ cal) {
    int t = blockIdx.x * blockDim.x + threadIdx.x;
    if (t >= T_STEPS) return;
    int wid = t / 288, sid = t % 288;
    float f[29];
    for (int i = 0; i < 8; ++i)  f[i] = wd_emb[wid * 8 + i];
    for (int i = 0; i < 16; ++i) f[8 + i] = sl_emb[sid * 16 + i];
    float ta = (6.28318530717958647692f / 288.f) * (float)sid;
    float wa = (6.28318530717958647692f / 7.f) * (float)wid;
    f[24] = __sinf(ta); f[25] = __cosf(ta);
    f[26] = __sinf(wa); f[27] = __cosf(wa);
    f[28] = (wid >= 5) ? 1.f : 0.f;
    float z[32];
    for (int j = 0; j < 32; ++j) {
        float a = bc1[j];
        for (int k = 0; k < 29; ++k) a += f[k] * Wc1[k * 32 + j];
        z[j] = a > 0.f ? a : 0.f;
    }
    for (int j = 0; j < 32; ++j) {
        float a = bc2[j];
        for (int k = 0; k < 32; ++k) a += z[k] * Wc2[k * 32 + j];
        cal[t * 32 + j] = a;
    }
}

// calrow[t][j] = cal[t] @ Wt1[32:64, :]   (node-invariant part of GEMM1)
__global__ void k_calrow(const float* __restrict__ cal, const float* __restrict__ Wt1,
                         float* __restrict__ calrow) {
    int tid = blockIdx.x * blockDim.x + threadIdx.x;
    if (tid >= T_STEPS * 256) return;
    int t = tid / 256, j = tid % 256;
    float acc = 0.f;
    for (int k = 0; k < 32; ++k) acc += cal[t * 32 + k] * Wt1[(32 + k) * 256 + j];
    calrow[tid] = acc;
}

// base[n][j] = xs[n] @ Wt1[0:32, :]       (time-invariant part of GEMM1)
__global__ void k_base(const float* __restrict__ xs, const float* __restrict__ Wt1,
                       float* __restrict__ base) {
    int tid = blockIdx.x * blockDim.x + threadIdx.x;
    if (tid >= N_NODES * 256) return;
    int n = tid / 256, j = tid % 256;
    float acc = 0.f;
    for (int k = 0; k < 32; ++k) acc += xs[n * 32 + k] * Wt1[k * 256 + j];
    base[tid] = acc;
}

__global__ void k_cvt_f16(const float* __restrict__ src, _Float16* __restrict__ dst,
                          int count, int srcOff) {
    int tid = blockIdx.x * blockDim.x + threadIdx.x;
    if (tid < count) dst[tid] = (_Float16)src[srcOff + tid];
}

// ---------------------------------------------------------------------------
// Fused temporal kernel: one block per time step.
//   TDM-stage weights -> pe MLP -> WMMA GEMM1 (pe@Wt1[64:96] + base + calrow)
//   -> GAT1 softmax/agg -> WMMA GEMM2 (y1@Wt2) -> GAT2 -> H_t -> head -> pred
// LDS: ~312 KB (f16 tiles), one workgroup per WGP.
// ---------------------------------------------------------------------------
__launch_bounds__(512, 1)
__global__ void k_temporal(
    const float* __restrict__ profile,
    const float* __restrict__ Wp1, const float* __restrict__ bp1,
    const float* __restrict__ Wp2, const float* __restrict__ bp2,
    const float* __restrict__ base, const float* __restrict__ calrow,
    const _Float16* __restrict__ wcomb,          // [32*256 | 64*64] f16
    const float* __restrict__ a1s, const float* __restrict__ a1d,
    const float* __restrict__ bt1,
    const float* __restrict__ a2s, const float* __restrict__ a2d,
    const float* __restrict__ bt2,
    const float* __restrict__ Wh1, const float* __restrict__ bh1,
    const float* __restrict__ Wh2, const float* __restrict__ bh2,
    const int* __restrict__ off, const int* __restrict__ lst,
    float* __restrict__ outH, float* __restrict__ outP) {

    const int t = blockIdx.x;
    const int tid = threadIdx.x;
    const int lane = tid & 31;
    const int wv = tid >> 5;

    extern __shared__ __align__(16) char smem[];
    _Float16* sPe = (_Float16*)smem;        // [400][32]  (reused after GEMM1)
    _Float16* sW1 = sPe + 400 * 32;         // [32][256]  (byte off 25600 = LDS_W_OFF)
    _Float16* sW2 = sW1 + 32 * 256;         // [64][64]   (contiguous with sW1)
    _Float16* sH  = sW2 + 64 * 64;          // [400][256] (h1) -> reused [400][64] (h2)
    _Float16* sY  = sH + 400 * 256;         // [400][64]  (y1) -> reused (H_t f16)
    float*    sAs = (float*)(sY + 400 * 64);// [400*4]
    float*    sAd = sAs + 1600;             // [400*4]
    // sPe region reuse after GEMM1:
    int*   sOff = (int*)sPe;                // [401]
    int*   sLst = sOff + 404;               // [3600]
    float* sWh1 = (float*)(sLst + 3600);    // [64*32]
    float* sBh1 = sWh1 + 2048;              // [32]
    float* sWh2 = sBh1 + 32;                // [32]

    // ---- Phase 0: TDM weight staging + prefetch + profile MLP -> sPe -------
    __builtin_prefetch(calrow + (size_t)t * 256, 0, 0);
    for (int i = tid; i < N_NODES; i += 512)
        __builtin_prefetch(base + (size_t)i * 256, 0, 0);

#if defined(__gfx1250__) && __has_builtin(__builtin_amdgcn_tensor_load_to_lds)
    if (wv == 0) {
        // Tensor DMA: one 1-D tile (12288 f16) global -> LDS @ sW1
        unsigned long long ga = (unsigned long long)(const void*)wcomb;
        v4u g0;
        g0[0] = 1u;                                   // count=1, user mode
        g0[1] = (unsigned)LDS_W_OFF;                  // lds_addr (bytes)
        g0[2] = (unsigned)(ga & 0xffffffffull);       // global_addr lo
        g0[3] = (unsigned)((ga >> 32) & 0x01ffffffull) | (2u << 30); // hi | type=2
        v8i g1;
        g1[0] = 1 << 16;                              // data_size = 2 bytes
        g1[1] = (int)(((unsigned)W_COMB_ELEMS & 0xffffu) << 16);       // tensor_dim0 lo16
        g1[2] = (int)((((unsigned)W_COMB_ELEMS >> 16) & 0xffffu) | (1u << 16)); // dim0 hi | tensor_dim1=1
        g1[3] = (int)(((unsigned)W_COMB_ELEMS & 0xffffu) << 16);       // tile_dim0
        g1[4] = 1;                                    // tile_dim1 = 1
        g1[5] = (int)W_COMB_ELEMS;                    // tensor_dim0_stride lo32
        g1[6] = 0;
        g1[7] = 0;
        v4i gz4 = {0, 0, 0, 0};
        v8i gz8 = {0, 0, 0, 0, 0, 0, 0, 0};
        __builtin_amdgcn_tensor_load_to_lds(g0, g1, gz4, gz4, gz8, 0);
        __builtin_amdgcn_s_wait_tensorcnt(0);
    }
#else
    for (int i = tid; i < W_COMB_ELEMS; i += 512) sW1[i] = wcomb[i];
#endif

    for (int n = tid; n < N_NODES; n += 512) {
        const float* p = profile + ((size_t)n * T_STEPS + t) * 8;
        float in[8];
        #pragma unroll
        for (int f = 0; f < 8; ++f) in[f] = p[f];
        float z[32];
        #pragma unroll
        for (int j = 0; j < 32; ++j) {
            float a = bp1[j];
            #pragma unroll
            for (int k = 0; k < 8; ++k) a += in[k] * Wp1[k * 32 + j];
            z[j] = a > 0.f ? a : 0.f;
        }
        for (int j = 0; j < 32; ++j) {
            float a = bp2[j];
            #pragma unroll
            for (int k = 0; k < 32; ++k) a += z[k] * Wp2[k * 32 + j];
            sPe[n * 32 + j] = (_Float16)a;
        }
    }
    __syncthreads();

    // ---- Phase 1: GEMM1 via WMMA: h = pe @ Wt1[64:96] + (base + calrow) ----
    {
        const int ni = wv;                     // 16 waves = 16 N tiles
        const int n0 = ni * 16;
        const int colB = n0 + (lane & 15);
        const int kbB = (lane >> 4) * 16;      // B: K = kbB + 2v + j
        v16h bfrag;
        #pragma unroll
        for (int v = 0; v < 8; ++v) {
            bfrag[2 * v]     = sW1[(kbB + 2 * v) * 256 + colB];
            bfrag[2 * v + 1] = sW1[(kbB + 2 * v + 1) * 256 + colB];
        }
        const int mlow = lane & 15;
        const int kbA = (lane >> 4) * 16;      // A: K = kbA..+15
        const int mD = (lane >> 4) << 3;       // C/D: m = r + mD
        const int nD = lane & 15;
        for (int mi = 0; mi < 25; ++mi) {
            const int m0 = mi * 16;
            const _Float16* ar = sPe + (m0 + mlow) * 32;
            v16h afrag;
            #pragma unroll
            for (int i = 0; i < 8; ++i) afrag[i] = ar[kbA + i];
            #pragma unroll
            for (int i = 0; i < 8; ++i) afrag[8 + i] = ar[kbA + 8 + i];
            const float crow = calrow[(size_t)t * 256 + n0 + nD];
            v8f acc;
            #pragma unroll
            for (int r = 0; r < 8; ++r)
                acc[r] = base[(size_t)(m0 + mD + r) * 256 + n0 + nD] + crow;
            acc = __builtin_amdgcn_wmma_f32_16x16x32_f16(
                false, afrag, false, bfrag, (short)0, acc, false, false);
            #pragma unroll
            for (int r = 0; r < 8; ++r)
                sH[(m0 + mD + r) * 256 + n0 + nD] = (_Float16)acc[r];
        }
    }
    __syncthreads();

    // ---- Phase 2: attention logits + stage CSR/head weights in LDS ---------
    for (int i = tid; i < 401; i += 512) sOff[i] = off[i];
    for (int i = tid; i < N_TOT; i += 512) sLst[i] = lst[i];
    for (int i = tid; i < 2048; i += 512) sWh1[i] = Wh1[i];
    if (tid < 32) { sBh1[tid] = bh1[tid]; sWh2[tid] = Wh2[tid]; }
    for (int i = tid; i < N_NODES * 4; i += 512) {
        int n = i >> 2, hh = i & 3;
        const _Float16* hr = sH + n * 256 + hh * 64;
        const float* vs = a1s + hh * 64;
        const float* vd = a1d + hh * 64;
        float s = 0.f, d = 0.f;
        #pragma unroll 8
        for (int c = 0; c < 64; ++c) {
            float v = (float)hr[c];
            s += v * vs[c];
            d += v * vd[c];
        }
        sAs[i] = s; sAd[i] = d;
    }
    __syncthreads();

    // ---- Phase 3: GAT1 softmax + aggregation (deterministic) ---------------
    for (int task = tid; task < N_NODES * 2; task += 512) {
        int n = task >> 1, cb = (task & 1) * 32;
        float acc[32];
        #pragma unroll
        for (int c = 0; c < 32; ++c) acc[c] = 0.f;
        int b = sOff[n], e = sOff[n + 1];
        for (int hh = 0; hh < 4; ++hh) {
            float ad = sAd[n * 4 + hh];
            float m = -1e30f;
            for (int k = b; k < e; ++k) {
                float v = sAs[sLst[k] * 4 + hh] + ad;
                v = v > 0.f ? v : 0.2f * v;
                m = fmaxf(m, v);
            }
            float den = 0.f;
            for (int k = b; k < e; ++k) {
                float v = sAs[sLst[k] * 4 + hh] + ad;
                v = v > 0.f ? v : 0.2f * v;
                den += __expf(v - m);
            }
            float inv = 1.f / den;
            for (int k = b; k < e; ++k) {
                float v = sAs[sLst[k] * 4 + hh] + ad;
                v = v > 0.f ? v : 0.2f * v;
                float w = __expf(v - m) * inv;
                const _Float16* hs = sH + sLst[k] * 256 + hh * 64 + cb;
                #pragma unroll
                for (int c = 0; c < 32; ++c) acc[c] += w * (float)hs[c];
            }
        }
        for (int c = 0; c < 32; ++c) {
            float y = acc[c] * 0.25f + bt1[cb + c];
            sY[n * 64 + cb + c] = (_Float16)(y > 0.f ? y : 0.f);
        }
    }
    __syncthreads();

    // ---- Phase 4: GEMM2 via WMMA: h2 = y1 @ Wt2 (K=64) ---------------------
    for (int tile = wv; tile < 100; tile += 16) {
        int m0 = (tile >> 2) * 16, n0 = (tile & 3) * 16;
        const int mlow = lane & 15;
        const int kbA = (lane >> 4) * 16;
        const int colB = n0 + (lane & 15);
        const int kbB = (lane >> 4) * 16;
        v8f acc;
        #pragma unroll
        for (int r = 0; r < 8; ++r) acc[r] = 0.f;
        #pragma unroll
        for (int ks = 0; ks < 2; ++ks) {
            const int k0 = ks * 32;
            const _Float16* ar = sY + (m0 + mlow) * 64 + k0;
            v16h afrag, bfrag;
            #pragma unroll
            for (int i = 0; i < 8; ++i) afrag[i] = ar[kbA + i];
            #pragma unroll
            for (int i = 0; i < 8; ++i) afrag[8 + i] = ar[kbA + 8 + i];
            #pragma unroll
            for (int v = 0; v < 8; ++v) {
                bfrag[2 * v]     = sW2[(k0 + kbB + 2 * v) * 64 + colB];
                bfrag[2 * v + 1] = sW2[(k0 + kbB + 2 * v + 1) * 64 + colB];
            }
            acc = __builtin_amdgcn_wmma_f32_16x16x32_f16(
                false, afrag, false, bfrag, (short)0, acc, false, false);
        }
        const int mD = (lane >> 4) << 3;
        const int nD = lane & 15;
        #pragma unroll
        for (int r = 0; r < 8; ++r)
            sH[(m0 + mD + r) * 64 + n0 + nD] = (_Float16)acc[r];  // sH reused: h2[400][64]
    }
    __syncthreads();

    // ---- Phase 5: attention logits layer 2 (1 head) ------------------------
    for (int n = tid; n < N_NODES; n += 512) {
        const _Float16* hr = sH + n * 64;
        float s = 0.f, d = 0.f;
        #pragma unroll 8
        for (int c = 0; c < 64; ++c) {
            float v = (float)hr[c];
            s += v * a2s[c];
            d += v * a2d[c];
        }
        sAs[n] = s; sAd[n] = d;
    }
    __syncthreads();

    // ---- Phase 6: GAT2 aggregation + relu -> H_t (b128 stores) + sY (f16) --
    for (int task = tid; task < N_NODES * 2; task += 512) {
        int n = task >> 1, cb = (task & 1) * 32;
        float acc[32];
        #pragma unroll
        for (int c = 0; c < 32; ++c) acc[c] = 0.f;
        int b = sOff[n], e = sOff[n + 1];
        float ad = sAd[n];
        float m = -1e30f;
        for (int k = b; k < e; ++k) {
            float v = sAs[sLst[k]] + ad;
            v = v > 0.f ? v : 0.2f * v;
            m = fmaxf(m, v);
        }
        float den = 0.f;
        for (int k = b; k < e; ++k) {
            float v = sAs[sLst[k]] + ad;
            v = v > 0.f ? v : 0.2f * v;
            den += __expf(v - m);
        }
        float inv = 1.f / den;
        for (int k = b; k < e; ++k) {
            float v = sAs[sLst[k]] + ad;
            v = v > 0.f ? v : 0.2f * v;
            float w = __expf(v - m) * inv;
            const _Float16* hs = sH + sLst[k] * 64 + cb;
            #pragma unroll
            for (int c = 0; c < 32; ++c) acc[c] += w * (float)hs[c];
        }
        float* dst = outH + ((size_t)t * N_NODES + n) * 64 + cb;
        #pragma unroll
        for (int q = 0; q < 8; ++q) {
            v4f v;
            #pragma unroll
            for (int j = 0; j < 4; ++j) {
                float y = acc[q * 4 + j] + bt2[cb + q * 4 + j];
                y = y > 0.f ? y : 0.f;
                v[j] = y;
                sY[n * 64 + cb + q * 4 + j] = (_Float16)y;
            }
            *(v4f*)(dst + q * 4) = v;
        }
    }
    __syncthreads();

    // ---- Phase 7: head MLP -> pred (weights from LDS) ----------------------
    for (int n = tid; n < N_NODES; n += 512) {
        const _Float16* hr = sY + n * 64;
        float p = bh2[0];
        for (int j = 0; j < 32; ++j) {
            float a = sBh1[j];
            #pragma unroll 8
            for (int k = 0; k < 64; ++k) a += (float)hr[k] * sWh1[k * 32 + j];
            a = a > 0.f ? a : 0.f;
            p += a * sWh2[j];
        }
        outP[(size_t)t * N_NODES + n] = p;
    }
}

// ---------------------------------------------------------------------------
extern "C" void kernel_launch(void* const* d_in, const int* in_sizes, int n_in,
                              void* d_out, int out_size, void* d_ws, size_t ws_size,
                              hipStream_t stream) {
    (void)in_sizes; (void)n_in; (void)out_size; (void)ws_size;
    const float* x_static = (const float*)d_in[0];
    const float* profile  = (const float*)d_in[1];
    const int*   edge_idx = (const int*)  d_in[2];
    const float* Ws1 = (const float*)d_in[3];
    const float* as1_src = (const float*)d_in[4];
    const float* as1_dst = (const float*)d_in[5];
    const float* bs1 = (const float*)d_in[6];
    const float* Ws2 = (const float*)d_in[7];
    const float* as2_src = (const float*)d_in[8];
    const float* as2_dst = (const float*)d_in[9];
    const float* bs2 = (const float*)d_in[10];
    const float* wd_emb = (const float*)d_in[11];
    const float* sl_emb = (const float*)d_in[12];
    const float* Wc1 = (const float*)d_in[13];
    const float* bc1 = (const float*)d_in[14];
    const float* Wc2 = (const float*)d_in[15];
    const float* bc2 = (const float*)d_in[16];
    const float* Wp1 = (const float*)d_in[17];
    const float* bp1 = (const float*)d_in[18];
    const float* Wp2 = (const float*)d_in[19];
    const float* bp2 = (const float*)d_in[20];
    const float* Wt1 = (const float*)d_in[21];
    const float* at1_src = (const float*)d_in[22];
    const float* at1_dst = (const float*)d_in[23];
    const float* bt1 = (const float*)d_in[24];
    const float* Wt2 = (const float*)d_in[25];
    const float* at2_src = (const float*)d_in[26];
    const float* at2_dst = (const float*)d_in[27];
    const float* bt2 = (const float*)d_in[28];
    const float* Wh1 = (const float*)d_in[29];
    const float* bh1 = (const float*)d_in[30];
    const float* Wh2 = (const float*)d_in[31];
    const float* bh2 = (const float*)d_in[32];

    // Workspace carve-out (256B aligned)
    size_t o = 0;
    auto take = [&](size_t bytes) {
        size_t r = o;
        o += (bytes + 255) & ~(size_t)255;
        return r;
    };
    char* ws = (char*)d_ws;
    int*   csr_off = (int*)  (ws + take(401 * 4));
    int*   csr_src = (int*)  (ws + take(N_TOT * 4));
    float* h1      = (float*)(ws + take((size_t)N_NODES * 128 * 4));
    float* als1    = (float*)(ws + take(N_NODES * 4 * 4));
    float* ald1    = (float*)(ws + take(N_NODES * 4 * 4));
    float* xs1     = (float*)(ws + take(N_NODES * 32 * 4));
    float* h2s     = (float*)(ws + take(N_NODES * 32 * 4));
    float* als2    = (float*)(ws + take(N_NODES * 4));
    float* ald2    = (float*)(ws + take(N_NODES * 4));
    float* xs      = (float*)(ws + take(N_NODES * 32 * 4));
    float* cal     = (float*)(ws + take((size_t)T_STEPS * 32 * 4));
    float* calrow  = (float*)(ws + take((size_t)T_STEPS * 256 * 4));
    float* base    = (float*)(ws + take((size_t)N_NODES * 256 * 4));
    _Float16* wcomb = (_Float16*)(ws + take((size_t)W_COMB_ELEMS * 2));

    float* outH = (float*)d_out;
    float* outP = outH + (size_t)T_STEPS * N_NODES * 64;

    // CSR (deterministic)
    k_csr<<<1, 1, 0, stream>>>(edge_idx, csr_off, csr_src);

    // Static GAT layer 1: 16 -> 4x32 -> mean -> 32
    k_dense<<<(N_NODES * 128 + 255) / 256, 256, 0, stream>>>(x_static, Ws1, h1, N_NODES, 16, 128);
    k_al<<<(N_NODES * 4 + 255) / 256, 256, 0, stream>>>(h1, as1_src, as1_dst, als1, ald1, N_NODES, 4, 32);
    k_gat_agg<<<(N_NODES + 63) / 64, 64, 0, stream>>>(h1, als1, ald1, csr_off, csr_src, bs1, xs1, N_NODES, 4, 32);
    // Static GAT layer 2: 32 -> 32
    k_dense<<<(N_NODES * 32 + 255) / 256, 256, 0, stream>>>(xs1, Ws2, h2s, N_NODES, 32, 32);
    k_al<<<(N_NODES + 255) / 256, 256, 0, stream>>>(h2s, as2_src, as2_dst, als2, ald2, N_NODES, 1, 32);
    k_gat_agg<<<(N_NODES + 63) / 64, 64, 0, stream>>>(h2s, als2, ald2, csr_off, csr_src, bs2, xs, N_NODES, 1, 32);

    // Calendar encoder + precomputed GEMM1 pieces
    k_cal<<<(T_STEPS + 127) / 128, 128, 0, stream>>>(wd_emb, sl_emb, Wc1, bc1, Wc2, bc2, cal);
    k_calrow<<<(T_STEPS * 256 + 255) / 256, 256, 0, stream>>>(cal, Wt1, calrow);
    k_base<<<(N_NODES * 256 + 255) / 256, 256, 0, stream>>>(xs, Wt1, base);

    // f16 weight conversion for WMMA (Wt1 rows 64..95 | Wt2), contiguous for TDM
    k_cvt_f16<<<(32 * 256 + 255) / 256, 256, 0, stream>>>(Wt1, wcomb, 32 * 256, 64 * 256);
    k_cvt_f16<<<(64 * 64 + 255) / 256, 256, 0, stream>>>(Wt2, wcomb + 32 * 256, 64 * 64, 0);

    // Fused temporal GAT, one block per time step, ~312 KB dynamic LDS
    constexpr size_t SMEM =
        (size_t)(400 * 32 + 32 * 256 + 64 * 64 + 400 * 256 + 400 * 64) * sizeof(_Float16) +
        (size_t)3200 * sizeof(float);
    (void)hipFuncSetAttribute((const void*)k_temporal,
                              hipFuncAttributeMaxDynamicSharedMemorySize, (int)SMEM);
    k_temporal<<<T_STEPS, 512, SMEM, stream>>>(
        profile, Wp1, bp1, Wp2, bp2, base, calrow,
        wcomb, at1_src, at1_dst, bt1,
        at2_src, at2_dst, bt2,
        Wh1, bh1, Wh2, bh2,
        csr_off, csr_src, outH, outP);
}
